// GCN_22686017257478
// MI455X (gfx1250) — matrix operand order repaired and checked by reading, not compiled, not run on hardware
//
#include <hip/hip_runtime.h>
#include <hip/hip_bf16.h>

typedef __bf16 bf16_t;
typedef __attribute__((ext_vector_type(16))) __bf16 v16bf;
typedef __attribute__((ext_vector_type(8)))  float  v8f;

#define DD 512   // feature dim (D == H == 512)

// ---------------- WMMA helpers ----------------

__device__ inline v16bf load_frag(const bf16_t* p0, const bf16_t* p1) {
  union { v16bf v; uint4 q[2]; } u;
  u.q[0] = *reinterpret_cast<const uint4*>(p0);
  u.q[1] = *reinterpret_cast<const uint4*>(p1);
  return u.v;
}

__device__ inline v8f wmma_bf16(v16bf a, v16bf b, v8f c) {
  // D = A(16x32 bf16) * B(32x16 bf16) + C(16x16 f32)
  return __builtin_amdgcn_wmma_f32_16x16x32_bf16(
      /*neg_a=*/false, a, /*neg_b=*/false, b,
      /*c_mod=*/(short)0, c, /*reuse_a=*/false, /*reuse_b=*/false);
}

// ---------------- generic bf16 GEMM:  C[M,512] = A[M,K] * Bt[512,K]^T ----------------
// Bt is the weight pre-transposed to [N,K] so B fragments are contiguous.
// Each wave computes a 32x32 tile via 2x2 wmma tiles. 8 waves / block.

template<bool RELU, bool HASBIAS, bool OUTF, bool OUTB>
__global__ __launch_bounds__(256) void gemm_bf16(
    const bf16_t* __restrict__ A, const bf16_t* __restrict__ Bt,
    const float* __restrict__ bias, float* __restrict__ Cf, bf16_t* __restrict__ Cb,
    int M, int Nc, int K, int tiles_m, int tiles_n)
{
  int tid  = threadIdx.x;
  int wg   = blockIdx.x * (blockDim.x >> 5) + (tid >> 5);
  if (wg >= tiles_m * tiles_n) return;          // wave-uniform exit
  int mt = wg / tiles_n, nt = wg % tiles_n;     // n fastest -> A rows stay hot
  int lane = tid & 31, lo = lane & 15, hi = lane >> 4;

  int ra0 = mt * 32 + lo;
  int ra1 = ra0 + 16;
  int ca0 = ra0 < M ? ra0 : M - 1;
  int ca1 = ra1 < M ? ra1 : M - 1;
  const bf16_t* A0 = A + (size_t)ca0 * K;
  const bf16_t* A1 = A + (size_t)ca1 * K;
  const bf16_t* B0 = Bt + (size_t)(nt * 32 + lo) * K + hi * 16;
  const bf16_t* B1 = B0 + (size_t)16 * K;

  v8f c00 = {}, c01 = {}, c10 = {}, c11 = {};
  for (int k0 = 0; k0 < K; k0 += 32) {
    v16bf a0 = load_frag(A0 + k0 + hi * 16, A0 + k0 + hi * 16 + 8);
    v16bf a1 = load_frag(A1 + k0 + hi * 16, A1 + k0 + hi * 16 + 8);
    v16bf b0 = load_frag(B0 + k0,          B0 + k0 + 8);
    v16bf b1 = load_frag(B1 + k0,          B1 + k0 + 8);
    c00 = wmma_bf16(a0, b0, c00);
    c01 = wmma_bf16(a0, b1, c01);
    c10 = wmma_bf16(a1, b0, c10);
    c11 = wmma_bf16(a1, b1, c11);
  }

  int colA = nt * 32 + lo;
  int colB = colA + 16;
  float bA = HASBIAS ? bias[colA] : 0.f;
  float bB = HASBIAS ? bias[colB] : 0.f;
#pragma unroll
  for (int r = 0; r < 8; ++r) {
    int row0 = mt * 32 + hi * 8 + r;   // rows of c00/c01
    int row1 = row0 + 16;              // rows of c10/c11
    if (row0 < M) {
      float v0 = c00[r] + bA, v1 = c01[r] + bB;
      if (RELU) { v0 = fmaxf(v0, 0.f); v1 = fmaxf(v1, 0.f); }
      size_t o0 = (size_t)row0 * Nc + colA, o1 = (size_t)row0 * Nc + colB;
      if (OUTF) { Cf[o0] = v0; Cf[o1] = v1; }
      if (OUTB) { Cb[o0] = (bf16_t)v0; Cb[o1] = (bf16_t)v1; }
    }
    if (row1 < M) {
      float v0 = c10[r] + bA, v1 = c11[r] + bB;
      if (RELU) { v0 = fmaxf(v0, 0.f); v1 = fmaxf(v1, 0.f); }
      size_t o0 = (size_t)row1 * Nc + colA, o1 = (size_t)row1 * Nc + colB;
      if (OUTF) { Cf[o0] = v0; Cf[o1] = v1; }
      if (OUTB) { Cb[o0] = (bf16_t)v0; Cb[o1] = (bf16_t)v1; }
    }
  }
}

// ---------------- head GEMM fused with final dot ----------------
// t = relu(p @ lin_in + b) ; out[row] += sum_col t[row,col] * w2[col]
__global__ __launch_bounds__(256) void head_gemm(
    const bf16_t* __restrict__ A, const bf16_t* __restrict__ Bt,
    const float* __restrict__ bias, const float* __restrict__ w2,
    float* __restrict__ out, int M, int K, int tiles_m, int tiles_n)
{
  int tid  = threadIdx.x;
  int wg   = blockIdx.x * (blockDim.x >> 5) + (tid >> 5);
  if (wg >= tiles_m * tiles_n) return;
  int mt = wg / tiles_n, nt = wg % tiles_n;
  int lane = tid & 31, lo = lane & 15, hi = lane >> 4;

  int ra0 = mt * 32 + lo;
  int ra1 = ra0 + 16;
  int ca0 = ra0 < M ? ra0 : M - 1;
  int ca1 = ra1 < M ? ra1 : M - 1;
  const bf16_t* A0 = A + (size_t)ca0 * K;
  const bf16_t* A1 = A + (size_t)ca1 * K;
  const bf16_t* B0 = Bt + (size_t)(nt * 32 + lo) * K + hi * 16;
  const bf16_t* B1 = B0 + (size_t)16 * K;

  v8f c00 = {}, c01 = {}, c10 = {}, c11 = {};
  for (int k0 = 0; k0 < K; k0 += 32) {
    v16bf a0 = load_frag(A0 + k0 + hi * 16, A0 + k0 + hi * 16 + 8);
    v16bf a1 = load_frag(A1 + k0 + hi * 16, A1 + k0 + hi * 16 + 8);
    v16bf b0 = load_frag(B0 + k0,          B0 + k0 + 8);
    v16bf b1 = load_frag(B1 + k0,          B1 + k0 + 8);
    c00 = wmma_bf16(a0, b0, c00);
    c01 = wmma_bf16(a0, b1, c01);
    c10 = wmma_bf16(a1, b0, c10);
    c11 = wmma_bf16(a1, b1, c11);
  }

  int colA = nt * 32 + lo, colB = colA + 16;
  float bA = bias[colA], bB = bias[colB];
  float wA = w2[colA],   wB = w2[colB];
#pragma unroll
  for (int r = 0; r < 8; ++r) {
    int row0 = mt * 32 + hi * 8 + r;
    int row1 = row0 + 16;
    float s0 = fmaxf(c00[r] + bA, 0.f) * wA + fmaxf(c01[r] + bB, 0.f) * wB;
    float s1 = fmaxf(c10[r] + bA, 0.f) * wA + fmaxf(c11[r] + bB, 0.f) * wB;
#pragma unroll
    for (int o = 1; o < 16; o <<= 1) {
      s0 += __shfl_xor(s0, o, 32);
      s1 += __shfl_xor(s1, o, 32);
    }
    if (lo == 0) {
      if (row0 < M) atomicAdd(&out[row0], s0);
      if (row1 < M) atomicAdd(&out[row1], s1);
    }
  }
}

// ---------------- elementwise / graph kernels ----------------

__global__ void k_f32_to_bf16(const float* __restrict__ in, bf16_t* __restrict__ out, int n) {
  int i = blockIdx.x * blockDim.x + threadIdx.x;
  if (i < n) out[i] = (bf16_t)in[i];
}

__global__ void k_transpose_w(const float* __restrict__ W, bf16_t* __restrict__ Wt, int K, int N) {
  int i = blockIdx.x * blockDim.x + threadIdx.x;
  if (i < K * N) {
    int k = i / N, n = i % N;
    Wt[(size_t)n * K + k] = (bf16_t)W[i];
  }
}

__global__ void k_zero_f32(float* __restrict__ p, int n) {
  int i = blockIdx.x * blockDim.x + threadIdx.x;
  if (i < n) p[i] = 0.f;
}

__global__ void k_deg(const int* __restrict__ dst, float* __restrict__ deg, int E) {
  int i = blockIdx.x * blockDim.x + threadIdx.x;
  if (i < E) atomicAdd(&deg[dst[i]], 1.0f);
}

__global__ void k_dinv(const float* __restrict__ deg, float* __restrict__ dinv, int n) {
  int i = blockIdx.x * blockDim.x + threadIdx.x;
  if (i < n) dinv[i] = rsqrtf(deg[i] + 1.0f);
}

// hsum[n,d] = h2[2n,d] + h2[2n+1,d]   (bf16 in, bf16 out, f32 math)
__global__ void k_sum_pairs(const bf16_t* __restrict__ h2, bf16_t* __restrict__ hs, int total) {
  int i = blockIdx.x * blockDim.x + threadIdx.x;
  if (i < total) {
    int n = i >> 9, d = i & (DD - 1);
    float a = (float)h2[(size_t)(2 * n) * DD + d];
    float b = (float)h2[(size_t)(2 * n + 1) * DD + d];
    hs[i] = (bf16_t)(a + b);
  }
}

// one block per edge: agg[dst] += h[src] * dinv[src]*dinv[dst]
__global__ __launch_bounds__(256) void k_scatter(
    const int* __restrict__ src, const int* __restrict__ dst,
    const float* __restrict__ h, const float* __restrict__ dinv,
    float* __restrict__ agg, int E)
{
  int e = blockIdx.x;
  if (e >= E) return;
  int s = src[e], t = dst[e];
  float norm = dinv[s] * dinv[t];
  const float* hs = h + (size_t)s * DD;
  float* at = agg + (size_t)t * DD;
  for (int d = threadIdx.x; d < DD; d += blockDim.x)
    atomicAdd(&at[d], hs[d] * norm);
}

// y1 = relu(agg + hh*dinv^2 + b)  -> bf16
__global__ void k_conv_finish1(const float* __restrict__ agg, const float* __restrict__ hh,
                               const float* __restrict__ dinv, const float* __restrict__ b,
                               bf16_t* __restrict__ y, int total) {
  int i = blockIdx.x * blockDim.x + threadIdx.x;
  if (i < total) {
    int n = i >> 9, d = i & (DD - 1);
    float di = dinv[n];
    float v = agg[i] + hh[i] * di * di + b[d];
    y[i] = (bf16_t)fmaxf(v, 0.f);
  }
}

// y2 = agg + hh*dinv^2 + b ; x1 = relu(x0+y2) ; emb = x1 + x0
__global__ void k_conv_finish2(const float* __restrict__ agg, const float* __restrict__ hh,
                               const float* __restrict__ dinv, const float* __restrict__ b,
                               const float* __restrict__ x0, float* __restrict__ emb, int total) {
  int i = blockIdx.x * blockDim.x + threadIdx.x;
  if (i < total) {
    int n = i >> 9, d = i & (DD - 1);
    float di = dinv[n];
    float v = agg[i] + hh[i] * di * di + b[d];
    float x0v = x0[i];
    float x1 = fmaxf(x0v + v, 0.f);
    emb[i] = x1 + x0v;
  }
}

// p[i,d] = emb[dst_idx[i],d] - emb[src_idx[i],d]  -> bf16
__global__ void k_pred_gather(const float* __restrict__ emb, const int* __restrict__ sidx,
                              const int* __restrict__ didx, bf16_t* __restrict__ p, int total) {
  int i = blockIdx.x * blockDim.x + threadIdx.x;
  if (i < total) {
    int row = i >> 9, d = i & (DD - 1);
    float v = emb[(size_t)didx[row] * DD + d] - emb[(size_t)sidx[row] * DD + d];
    p[i] = (bf16_t)v;
  }
}

__global__ void k_init_out(float* __restrict__ out, const float* __restrict__ b, int n) {
  int i = blockIdx.x * blockDim.x + threadIdx.x;
  if (i < n) out[i] = b[0];
}

// ---------------- launch ----------------

extern "C" void kernel_launch(void* const* d_in, const int* in_sizes, int n_in,
                              void* d_out, int out_size, void* d_ws, size_t ws_size,
                              hipStream_t stream) {
  const float* x        = (const float*)d_in[0];
  const int*   ei       = (const int*)  d_in[1];
  const int*   src_idx  = (const int*)  d_in[2];
  const int*   dst_idx  = (const int*)  d_in[3];
  const float* emb_in_W = (const float*)d_in[4];
  const float* emb_in_b = (const float*)d_in[5];
  const float* emb_out_W= (const float*)d_in[6];
  const float* emb_out_b= (const float*)d_in[7];
  const float* rho_W    = (const float*)d_in[8];
  const float* rho_b    = (const float*)d_in[9];
  const float* conv1_W  = (const float*)d_in[10];
  const float* conv1_b  = (const float*)d_in[11];
  const float* conv2_W  = (const float*)d_in[12];
  const float* conv2_b  = (const float*)d_in[13];
  const float* lin_in_W = (const float*)d_in[14];
  const float* lin_in_b = (const float*)d_in[15];
  const float* lin_out_W= (const float*)d_in[16];
  const float* lin_out_b= (const float*)d_in[17];
  float* out = (float*)d_out;

  const int E  = in_sizes[1] / 2;
  const int P  = in_sizes[2];
  const int N  = in_sizes[0] / (2 * DD);
  const int M2 = 2 * N;

  // workspace carve (256B-aligned)
  size_t off = 0;
  char* base = (char*)d_ws;
  auto alloc = [&](size_t bytes) -> void* {
    void* p = base + off;
    off = (off + bytes + 255) & ~(size_t)255;
    return p;
  };
  bf16_t* WtEmbIn  = (bf16_t*)alloc((size_t)DD * DD * 2);
  bf16_t* WtEmbOut = (bf16_t*)alloc((size_t)DD * DD * 2);
  bf16_t* WtRho    = (bf16_t*)alloc((size_t)DD * DD * 2);
  bf16_t* WtC1     = (bf16_t*)alloc((size_t)DD * DD * 2);
  bf16_t* WtC2     = (bf16_t*)alloc((size_t)DD * DD * 2);
  bf16_t* WtLin    = (bf16_t*)alloc((size_t)DD * DD * 2);
  bf16_t* xbf      = (bf16_t*)alloc((size_t)M2 * DD * 2);
  bf16_t* h1bf     = (bf16_t*)alloc((size_t)M2 * DD * 2);
  bf16_t* h2bf     = (bf16_t*)alloc((size_t)M2 * DD * 2);
  bf16_t* hsumbf   = (bf16_t*)alloc((size_t)N * DD * 2);
  float*  x0f      = (float*) alloc((size_t)N * DD * 4);
  bf16_t* x0bf     = (bf16_t*)alloc((size_t)N * DD * 2);
  float*  hhf      = (float*) alloc((size_t)N * DD * 4);
  float*  aggf     = (float*) alloc((size_t)N * DD * 4);
  bf16_t* y1bf     = (bf16_t*)alloc((size_t)N * DD * 2);
  float*  embf     = (float*) alloc((size_t)N * DD * 4);
  float*  deg      = (float*) alloc((size_t)N * 4);
  float*  dinv     = (float*) alloc((size_t)N * 4);
  bf16_t* pbf      = (bf16_t*)alloc((size_t)P * DD * 2);

  const int B = 256;
  const int wn = DD * DD;
  const int wgrid = (wn + B - 1) / B;
  k_transpose_w<<<wgrid, B, 0, stream>>>(emb_in_W,  WtEmbIn,  DD, DD);
  k_transpose_w<<<wgrid, B, 0, stream>>>(emb_out_W, WtEmbOut, DD, DD);
  k_transpose_w<<<wgrid, B, 0, stream>>>(rho_W,     WtRho,    DD, DD);
  k_transpose_w<<<wgrid, B, 0, stream>>>(conv1_W,   WtC1,     DD, DD);
  k_transpose_w<<<wgrid, B, 0, stream>>>(conv2_W,   WtC2,     DD, DD);
  k_transpose_w<<<wgrid, B, 0, stream>>>(lin_in_W,  WtLin,    DD, DD);

  const int nx = M2 * DD;
  k_f32_to_bf16<<<(nx + B - 1) / B, B, 0, stream>>>(x, xbf, nx);

  const int TN = DD / 32;  // 16 column tiles
  auto gemm_grid = [&](int tm) { return (tm * TN + 7) / 8; };

  // 1) h1 = relu(x @ emb_in_W + b)
  int tm2 = (M2 + 31) / 32;
  gemm_bf16<true, true, false, true><<<gemm_grid(tm2), B, 0, stream>>>(
      xbf, WtEmbIn, emb_in_b, nullptr, h1bf, M2, DD, DD, tm2, TN);
  // 2) h2 = h1 @ emb_out_W + b
  gemm_bf16<false, true, false, true><<<gemm_grid(tm2), B, 0, stream>>>(
      h1bf, WtEmbOut, emb_out_b, nullptr, h2bf, M2, DD, DD, tm2, TN);
  // 3) sum over set dim
  const int nN = N * DD;
  k_sum_pairs<<<(nN + B - 1) / B, B, 0, stream>>>(h2bf, hsumbf, nN);
  // 4) x0 = relu(hsum @ rho_W + b)   (f32 + bf16 outputs)
  int tmN = (N + 31) / 32;
  gemm_bf16<true, true, true, true><<<gemm_grid(tmN), B, 0, stream>>>(
      hsumbf, WtRho, rho_b, x0f, x0bf, N, DD, DD, tmN, TN);

  // degrees (shared by both convs)
  const int* esrc = ei;
  const int* edst = ei + E;
  k_zero_f32<<<(N + B - 1) / B, B, 0, stream>>>(deg, N);
  k_deg<<<(E + B - 1) / B, B, 0, stream>>>(edst, deg, E);
  k_dinv<<<(N + B - 1) / B, B, 0, stream>>>(deg, dinv, N);

  // conv1
  gemm_bf16<false, false, true, false><<<gemm_grid(tmN), B, 0, stream>>>(
      x0bf, WtC1, nullptr, hhf, nullptr, N, DD, DD, tmN, TN);
  k_zero_f32<<<(nN + B - 1) / B, B, 0, stream>>>(aggf, nN);
  k_scatter<<<E, B, 0, stream>>>(esrc, edst, hhf, dinv, aggf, E);
  k_conv_finish1<<<(nN + B - 1) / B, B, 0, stream>>>(aggf, hhf, dinv, conv1_b, y1bf, nN);

  // conv2
  gemm_bf16<false, false, true, false><<<gemm_grid(tmN), B, 0, stream>>>(
      y1bf, WtC2, nullptr, hhf, nullptr, N, DD, DD, tmN, TN);
  k_zero_f32<<<(nN + B - 1) / B, B, 0, stream>>>(aggf, nN);
  k_scatter<<<E, B, 0, stream>>>(esrc, edst, hhf, dinv, aggf, E);
  k_conv_finish2<<<(nN + B - 1) / B, B, 0, stream>>>(aggf, hhf, dinv, conv2_b, x0f, embf, nN);

  // prediction head
  const int nP = P * DD;
  k_pred_gather<<<(nP + B - 1) / B, B, 0, stream>>>(embf, src_idx, dst_idx, pbf, nP);
  k_init_out<<<(P + B - 1) / B, B, 0, stream>>>(out, lin_out_b, P);
  int tmP = (P + 31) / 32;
  head_gemm<<<(tmP * TN + 7) / 8, B, 0, stream>>>(
      pbf, WtLin, lin_in_b, lin_out_W, out, P, DD, tmP, TN);
}
